// SimpleModel_61074434949815
// MI455X (gfx1250) — matrix operand order repaired and checked by reading, not compiled, not run on hardware
//
#include <hip/hip_runtime.h>

typedef __attribute__((ext_vector_type(16))) _Float16 v16h;
typedef __attribute__((ext_vector_type(8)))  _Float16 v8h;
typedef __attribute__((ext_vector_type(8)))  float    v8f;

#define ROW   40      // halves per LDS row (80 B: keeps 16B alignment for b128 loads)
#define WAVES 8       // waves per 256-thread block (wave32)
#define TPW   8       // 16-sample tiles per wave

// ---------------------------------------------------------------------------
// Prep: pack 9 weight matrices (zero-padded to 32xN) into f16 WMMA B-operand
// lane layout. B(32x16,f16) layout used by this code (contiguous K, same order as A):
//   lane = 16*(K>=16) + N ;  half t (0..15) within lane holds K = 16*(lane>>4)+t
// ws layout: ws[j*512 + lane*16 + t]  -> each lane loads a contiguous v16h.
// ---------------------------------------------------------------------------
__global__ void prep_weights(const float* W0a, const float* W0b,
                             const float* W1a, const float* W1b,
                             const float* W2a, const float* W2b,
                             const float* W3a, const float* W3b,
                             const float* Wl1, _Float16* ws) {
  const int j = blockIdx.x;
  const float* W; int Kr, Nc;
  switch (j) {
    case 0: W = W0a; Kr = 12; Nc = 16; break;
    case 1: W = W0b; Kr = 16; Nc = 16; break;
    case 2: W = W1a; Kr = 8;  Nc = 16; break;
    case 3: W = W1b; Kr = 16; Nc = 16; break;
    case 4: W = W2a; Kr = 16; Nc = 16; break;
    case 5: W = W2b; Kr = 16; Nc = 16; break;
    case 6: W = W3a; Kr = 24; Nc = 16; break;
    case 7: W = W3b; Kr = 16; Nc = 16; break;
    default: W = Wl1; Kr = 32; Nc = 8;  break;
  }
  const int lane = threadIdx.x & 31;
  const int g = lane >> 4, n = lane & 15;
#pragma unroll
  for (int t = 0; t < 16; ++t) {
    const int K = 16 * g + t;
    const float v = (K < Kr && n < Nc) ? W[K * Nc + n] : 0.0f;
    ws[j * 512 + lane * 16 + t] = (_Float16)v;
  }
}

static __device__ __forceinline__ v8f wmma32(v16h a, v16h b, v8f c) {
  // D = A(16x32 f16) x B(32x16 f16) + C(16x16 f32)
  return __builtin_amdgcn_wmma_f32_16x16x32_f16(false, a, false, b,
                                                (short)0, c, false, false);
}

// A(16x32,f16) lane layout: lane m(+16g) holds row M=m;
// halves 0..15 -> K = 16g..16g+15 (matches packed B).
// Row-major [M][K] LDS tile -> two ds_load_b128 per lane.
static __device__ __forceinline__ v16h loadA(const _Float16* base, int m, int g) {
  union { v16h v; v8h h[2]; } u;
  const _Float16* r = base + m * ROW;
  u.h[0] = *(const v8h*)(r + 16 * g);
  u.h[1] = *(const v8h*)(r + 16 * g + 8);
  return u.v;
}

// per-lane 8-wide material embedding into an LDS row
#define EMB8(AMT, FAM, PER, EL, OFF)                                          \
  do {                                                                        \
    rowp[(OFF) + 0] = (_Float16)((AMT[s] - 1.0f) * (1.0f / 259.0f));          \
    const int f3_ = FAM[s] * 3;                                               \
    rowp[(OFF) + 1] = (_Float16)fam_emb[f3_ + 0];                             \
    rowp[(OFF) + 2] = (_Float16)fam_emb[f3_ + 1];                             \
    rowp[(OFF) + 3] = (_Float16)fam_emb[f3_ + 2];                             \
    const int p3_ = PER[s] * 3;                                               \
    rowp[(OFF) + 4] = (_Float16)per_emb[p3_ + 0];                             \
    rowp[(OFF) + 5] = (_Float16)per_emb[p3_ + 1];                             \
    rowp[(OFF) + 6] = (_Float16)per_emb[p3_ + 2];                             \
    rowp[(OFF) + 7] = (_Float16)((EL[s] - 0.9f) * (1.0f / 1.9f));             \
  } while (0)

__global__ __launch_bounds__(256)
void fused_mlp(const float* sup_amount, const int* sup_fam, const int* sup_per, const float* sup_elec,
               const float* m1_amount,  const int* m1_fam,  const int* m1_per,  const float* m1_elec,
               const float* m2_amount,  const int* m2_fam,  const int* m2_per,  const float* m2_elec,
               const float* m3_amount,  const int* m3_fam,  const int* m3_per,  const float* m3_elec,
               const int* branch_id, const float* temperature, const float* pch4,
               const float* po2, const float* par,
               const _Float16* wsW,
               const float* b0a, const float* b0b, const float* b1a, const float* b1b,
               const float* b2a, const float* b2b, const float* b3a, const float* b3b,
               const float* bl1,
               const float* Wl2, const float* bl2, const float* Wout, const float* bout,
               const float* fam_emb, const float* per_emb,
               float* out, int nTotal, int nTiles) {
  __shared__ __align__(16) _Float16 sm[WAVES * 1280];   // per wave: 2 tiles of 16 rows x ROW halves
  __shared__ int sbid[WAVES * 16];

  const int lane = threadIdx.x & 31;
  const int w    = threadIdx.x >> 5;
  const int g    = lane >> 4;       // lane group (K half for A, M group for C)
  const int mI   = lane & 15;       // row index for A loads / column N for C
  _Float16* r0 = sm + w * 1280;     // staging region 0 (x0 / hidden)
  _Float16* r1 = r0 + 640;          // staging region 1 (m1|m2|m3)

  // zero wave-private LDS once: A padding must be finite (NaN*0 = NaN in WMMA)
  for (int i = lane; i < 1280; i += 32) r0[i] = (_Float16)0.0f;

  // resident B operands: 9 matrices x v16h (8 VGPRs each)
  const v16h* wsv = (const v16h*)wsW;
  v16h Bw[9];
#pragma unroll
  for (int j = 0; j < 9; ++j) Bw[j] = wsv[j * 32 + lane];

  // biases: C layout gives every lane N = lane&15 for all 8 accumulators
  float bb[9];
  bb[0] = b0a[mI]; bb[1] = b0b[mI]; bb[2] = b1a[mI]; bb[3] = b1b[mI];
  bb[4] = b2a[mI]; bb[5] = b2b[mI]; bb[6] = b3a[mI]; bb[7] = b3b[mI];
  bb[8] = (mI < 8) ? bl1[mI] : 0.0f;

  const int waveG = blockIdx.x * WAVES + w;

  auto storeHiddenRelu = [&](v8f c, float bias) {
#pragma unroll
    for (int i = 0; i < 8; ++i) {
      float v = c[i] + bias;
      v = v > 0.0f ? v : 0.0f;
      r0[(i + 8 * g) * ROW + mI] = (_Float16)v;   // (M = i+8g, K-col = N = mI)
    }
  };

  for (int it = 0; it < TPW; ++it) {
    const int tile = waveG * TPW + it;            // wave-uniform
    if (tile >= nTiles) break;
    const int sbase = tile * 16;

    // ---- build feature rows (f16) in LDS -------------------------------
    if (lane < 16) {                               // x0 = [sup(8) | infos(4)]
      int s = sbase + lane; if (s >= nTotal) s = nTotal - 1;
      _Float16* rowp = r0 + lane * ROW;
      EMB8(sup_amount, sup_fam, sup_per, sup_elec, 0);
      rowp[8]  = (_Float16)((temperature[s] - 600.0f) * (1.0f / 300.0f));
      rowp[9]  = (_Float16)pch4[s];
      rowp[10] = (_Float16)po2[s];
      rowp[11] = (_Float16)par[s];
      sbid[w * 16 + lane] = branch_id[s];
    } else {                                       // [m1(8) | m2(8) | m3(8)]
      const int m = lane - 16;
      int s = sbase + m; if (s >= nTotal) s = nTotal - 1;
      _Float16* rowp = r1 + m * ROW;
      EMB8(m1_amount, m1_fam, m1_per, m1_elec, 0);
      EMB8(m2_amount, m2_fam, m2_per, m2_elec, 8);
      EMB8(m3_amount, m3_fam, m3_per, m3_elec, 16);
    }

    const v16h Ax0 = loadA(r0, mI, g);
    const v16h Am  = loadA(r1, mI, g);
    const v8f cz = {};

    // ---- base = relu(x0@W0a+b0a)@W0b+b0b -------------------------------
    storeHiddenRelu(wmma32(Ax0, Bw[0], cz), bb[0]);
    v16h Ah = loadA(r0, mI, g);
    v8f cbase = wmma32(Ah, Bw[1], cz);
#pragma unroll
    for (int i = 0; i < 8; ++i) cbase[i] += bb[1];

    // ---- f1 / f2 / f3 (shared A tile; zero-padded weight rows) ---------
    storeHiddenRelu(wmma32(Am, Bw[2], cz), bb[2]);
    Ah = loadA(r0, mI, g);
    v8f f1 = wmma32(Ah, Bw[3], cz);
#pragma unroll
    for (int i = 0; i < 8; ++i) f1[i] += bb[3];

    storeHiddenRelu(wmma32(Am, Bw[4], cz), bb[4]);
    Ah = loadA(r0, mI, g);
    v8f f2 = wmma32(Ah, Bw[5], cz);
#pragma unroll
    for (int i = 0; i < 8; ++i) f2[i] += bb[5];

    storeHiddenRelu(wmma32(Am, Bw[6], cz), bb[6]);
    Ah = loadA(r0, mI, g);
    v8f f3 = wmma32(Ah, Bw[7], cz);
#pragma unroll
    for (int i = 0; i < 8; ++i) f3[i] += bb[7];

    // ---- branch select + concat [front | base] -> Wl1 ------------------
#pragma unroll
    for (int i = 0; i < 8; ++i) {
      const int bid = sbid[w * 16 + i + 8 * g];    // branch id of row M=i+8g
      const float fr = (bid == 0) ? f1[i] : ((bid == 1) ? f2[i] : f3[i]);
      r0[(i + 8 * g) * ROW + mI]      = (_Float16)fr;
      r0[(i + 8 * g) * ROW + 16 + mI] = (_Float16)cbase[i];
    }
    Ah = loadA(r0, mI, g);
    v8f cl1 = wmma32(Ah, Bw[8], cz);               // K=32 real, N>=8 cols are 0
#pragma unroll
    for (int i = 0; i < 8; ++i) {
      float v = cl1[i] + bb[8];
      v = v > 0.0f ? v : 0.0f;
      if (mI < 8) r0[(i + 8 * g) * ROW + mI] = (_Float16)v;
    }

    // ---- tiny tail 8->4->1 on VALU (uniform scalar weights) ------------
    if (lane < 16) {
      float h1[8];
#pragma unroll
      for (int k = 0; k < 8; ++k) h1[k] = (float)r0[lane * ROW + k];
      float o = bout[0];
#pragma unroll
      for (int j = 0; j < 4; ++j) {
        float sacc = bl2[j];
#pragma unroll
        for (int k = 0; k < 8; ++k) sacc += h1[k] * Wl2[k * 4 + j];
        sacc = sacc > 0.0f ? sacc : 0.0f;
        o += sacc * Wout[j];
      }
      const int s = sbase + lane;
      if (s < nTotal) out[s] = o;
    }
  }
}

extern "C" void kernel_launch(void* const* d_in, const int* in_sizes, int n_in,
                              void* d_out, int out_size, void* d_ws, size_t ws_size,
                              hipStream_t stream) {
  (void)n_in; (void)out_size; (void)ws_size;
  const float* sup_amount  = (const float*)d_in[0];
  const int*   sup_fam     = (const int*)  d_in[1];
  const int*   sup_per     = (const int*)  d_in[2];
  const float* sup_elec    = (const float*)d_in[3];
  const float* m1_amount   = (const float*)d_in[4];
  const int*   m1_fam      = (const int*)  d_in[5];
  const int*   m1_per      = (const int*)  d_in[6];
  const float* m1_elec     = (const float*)d_in[7];
  const float* m2_amount   = (const float*)d_in[8];
  const int*   m2_fam      = (const int*)  d_in[9];
  const int*   m2_per      = (const int*)  d_in[10];
  const float* m2_elec     = (const float*)d_in[11];
  const float* m3_amount   = (const float*)d_in[12];
  const int*   m3_fam      = (const int*)  d_in[13];
  const int*   m3_per      = (const int*)  d_in[14];
  const float* m3_elec     = (const float*)d_in[15];
  const int*   branch_id   = (const int*)  d_in[16];
  const float* temperature = (const float*)d_in[17];
  const float* pch4        = (const float*)d_in[18];
  const float* po2         = (const float*)d_in[19];
  const float* par         = (const float*)d_in[20];
  const float* W0a = (const float*)d_in[21]; const float* b0a = (const float*)d_in[22];
  const float* W0b = (const float*)d_in[23]; const float* b0b = (const float*)d_in[24];
  const float* W1a = (const float*)d_in[25]; const float* b1a = (const float*)d_in[26];
  const float* W1b = (const float*)d_in[27]; const float* b1b = (const float*)d_in[28];
  const float* W2a = (const float*)d_in[29]; const float* b2a = (const float*)d_in[30];
  const float* W2b = (const float*)d_in[31]; const float* b2b = (const float*)d_in[32];
  const float* W3a = (const float*)d_in[33]; const float* b3a = (const float*)d_in[34];
  const float* W3b = (const float*)d_in[35]; const float* b3b = (const float*)d_in[36];
  const float* Wl1 = (const float*)d_in[37]; const float* bl1 = (const float*)d_in[38];
  const float* Wl2 = (const float*)d_in[39]; const float* bl2 = (const float*)d_in[40];
  const float* Wout = (const float*)d_in[41]; const float* bout = (const float*)d_in[42];
  const float* fam_emb = (const float*)d_in[43];
  const float* per_emb = (const float*)d_in[44];

  _Float16* ws = (_Float16*)d_ws;              // 9*512*2 = 9216 bytes used
  float* out = (float*)d_out;

  const int n = in_sizes[0];                   // B = 1048576
  const int nTiles = (n + 15) / 16;
  const int wavesTotal = (nTiles + TPW - 1) / TPW;
  const int blocks = (wavesTotal + WAVES - 1) / WAVES;

  prep_weights<<<9, 32, 0, stream>>>(W0a, W0b, W1a, W1b, W2a, W2b, W3a, W3b, Wl1, ws);
  fused_mlp<<<blocks, WAVES * 32, 0, stream>>>(
      sup_amount, sup_fam, sup_per, sup_elec,
      m1_amount, m1_fam, m1_per, m1_elec,
      m2_amount, m2_fam, m2_per, m2_elec,
      m3_amount, m3_fam, m3_per, m3_elec,
      branch_id, temperature, pch4, po2, par,
      ws,
      b0a, b0b, b1a, b1b, b2a, b2b, b3a, b3b, bl1,
      Wl2, bl2, Wout, bout,
      fam_emb, per_emb,
      out, n, nTiles);
}
